// VoxelBackBone8x_WA_37228776521771
// MI455X (gfx1250) — matrix operand order repaired and checked by reading, not compiled, not run on hardware
//
#include <hip/hip_runtime.h>

// ---------------------------------------------------------------------------
// Types: ext_vector types are trivially copyable (union-legal) and match the
// gfx1250 WMMA builtin signatures.
// ---------------------------------------------------------------------------
typedef __bf16 bf16;
typedef __attribute__((ext_vector_type(16))) __bf16    v16bf;
typedef __attribute__((ext_vector_type(8)))  float     v8f;
typedef __attribute__((ext_vector_type(4)))  unsigned  u32x4;

union ABfrag { u32x4 u[2]; v16bf v; };

#define WAVES 8
#define TPB   (WAVES * 32)

// ---------------------------------------------------------------------------
// Scatter voxel features into dense bf16 NDHWC level-0 tensor + mask0
// ---------------------------------------------------------------------------
__global__ void scatter_kernel(const float* __restrict__ vf,
                               const int* __restrict__ vc,
                               bf16* __restrict__ x0,
                               float* __restrict__ m0, int N_) {
  int i = blockIdx.x * blockDim.x + threadIdx.x;
  if (i >= N_) return;
  int b = vc[4 * i + 0], z = vc[4 * i + 1], y = vc[4 * i + 2], x = vc[4 * i + 3];
  int pos = ((b * 33 + z) * 160 + y) * 160 + x;
  m0[pos] = 1.0f;
  bf16* p = x0 + (size_t)pos * 16;  // channels padded 4 -> 16 (rest memset 0)
  p[0] = (bf16)vf[4 * i + 0];
  p[1] = (bf16)vf[4 * i + 1];
  p[2] = (bf16)vf[4 * i + 2];
  p[3] = (bf16)vf[4 * i + 3];
}

// ---------------------------------------------------------------------------
// down_mask: out = (conv(mask, ones, stride, pad) > 0).  Compile-time dims.
// ---------------------------------------------------------------------------
template <int DI, int HI, int WI, int DO, int HO, int WO,
          int KD, int KH, int KW, int SD, int SH, int SW, int PD, int PH, int PW>
__global__ void mask_down_kernel(const float* __restrict__ mi, float* __restrict__ mo) {
  constexpr int NUMM = 2 * DO * HO * WO;
  int i = blockIdx.x * blockDim.x + threadIdx.x;
  if (i >= NUMM) return;
  int ow = i % WO; int t = i / WO;
  int oh = t % HO; t /= HO;
  int od = t % DO; int ob = t / DO;
  float acc = 0.f;
#pragma unroll
  for (int kd = 0; kd < KD; ++kd) {
    int id = od * SD - PD + kd; if (id < 0 || id >= DI) continue;
#pragma unroll
    for (int kh = 0; kh < KH; ++kh) {
      int ih = oh * SH - PH + kh; if (ih < 0 || ih >= HI) continue;
#pragma unroll
      for (int kw = 0; kw < KW; ++kw) {
        int iw = ow * SW - PW + kw; if (iw < 0 || iw >= WI) continue;
        acc += mi[((ob * DI + id) * HI + ih) * WI + iw];
      }
    }
  }
  mo[i] = (acc > 0.f) ? 1.f : 0.f;
}

// ---------------------------------------------------------------------------
// Weight prep: OIDHW f32 -> bf16 B-fragment tiles
// layout: [ntile][tap][kslice][n(16)][k(32)], k zero-padded past Cin
// ---------------------------------------------------------------------------
template <int COUT, int CIN, int KD, int KH, int KW, int KSL>
__global__ void prep_w_kernel(const float* __restrict__ w, bf16* __restrict__ o) {
  constexpr int TAPS = KD * KH * KW;
  constexpr int TOTAL = (COUT / 16) * TAPS * KSL * 512;
  int i = blockIdx.x * blockDim.x + threadIdx.x;
  if (i >= TOTAL) return;
  int k = i & 31;
  int n = (i >> 5) & 15;
  int j = i >> 9;
  int ks = j % KSL; j /= KSL;
  int tap = j % TAPS;
  int nt = j / TAPS;
  int kw = tap % KW; int r = tap / KW; int kh = r % KH; int kd = r / KH;
  int co = nt * 16 + n;
  int ci = ks * 32 + k;
  float v = 0.f;
  if (ci < CIN) v = w[(((co * CIN + ci) * KD + kd) * KH + kh) * KW + kw];
  o[i] = (bf16)v;
}

// ---------------------------------------------------------------------------
// Implicit-GEMM masked conv3d + fused BN/ReLU/mask.  bf16 WMMA, f32 accum.
//   grid.x : M tiles (output voxels), 128/block = 8 waves x 16
//   grid.y : groups of NT out-channel tiles
// B-tiles staged into LDS via CDNA5 async global->LDS copies (ASYNCcnt),
// no VGPR round-trip; single s_wait_asynccnt before the barrier.
// Each wave loads each A fragment ONCE per (tap,kslice) and reuses it for
// NT WMMAs.  All geometry is compile-time -> div/mod become mul-shift.
// ---------------------------------------------------------------------------
template <int DIN, int HIN, int WIN, int DOUT, int HOUT, int WOUT,
          int CSIN, int CSOUT, int KSL, int NT,
          int KD, int KH, int KW, int SD, int SH, int SW,
          int PD, int PH, int PW, int FINAL>
__global__ __launch_bounds__(TPB) void conv_wmma_kernel(
    const bf16* __restrict__ in, bf16* __restrict__ out, float* __restrict__ fout,
    const bf16* __restrict__ wprep,
    const float* __restrict__ bng, const float* __restrict__ bnb,
    const float* __restrict__ bnm, const float* __restrict__ bnv,
    const float* __restrict__ maskOut) {
  constexpr int TAPS = KD * KH * KW;
  constexpr int NUMM = 2 * DOUT * HOUT * WOUT;
  __shared__ alignas(16) bf16 sB[NT * TAPS * KSL * 512];

  const int tid  = threadIdx.x;
  const int lane = tid & 31;
  const int wave = tid >> 5;
  const int ng   = blockIdx.y;  // group of NT consecutive n-tiles

  // --- Cooperative B-tile staging: async global -> LDS (16 B per lane) -----
  {
    const char* gsrc = (const char*)(wprep + (size_t)ng * NT * TAPS * KSL * 512);
    const unsigned ldsBase = (unsigned)(uintptr_t)(&sB[0]);  // low 32 bits = LDS byte addr
    constexpr int N16 = NT * TAPS * KSL * 64;  // 16-byte chunks
    for (int i = tid; i < N16; i += TPB) {
      unsigned      lds_addr = ldsBase + (unsigned)i * 16u;
      unsigned long gaddr    = (unsigned long)(uintptr_t)(gsrc + (size_t)i * 16);
      asm volatile("global_load_async_to_lds_b128 %0, %1, off"
                   :
                   : "v"(lds_addr), "v"(gaddr)
                   : "memory");
    }
    asm volatile("s_wait_asynccnt 0x0" ::: "memory");
  }
  __syncthreads();

  const int mBase = blockIdx.x * (16 * WAVES) + wave * 16;
  const int m    = lane & 15;
  const int half = lane >> 4;

  // A-fragment row for this lane (M = lane & 15)
  int posA = mBase + m;
  const bool posOK = posA < NUMM;
  if (!posOK) posA = 0;
  int ow = posA % WOUT; int t = posA / WOUT;
  int oh = t % HOUT; t /= HOUT;
  int od = t % DOUT; const int ob = t / DOUT;

  v8f acc[NT];
#pragma unroll
  for (int nt = 0; nt < NT; ++nt) acc[nt] = (v8f)0.f;
  const u32x4 z4 = (u32x4)0u;

  for (int tp = 0; tp < TAPS; ++tp) {
    const int kw = tp % KW; const int r = tp / KW;
    const int kh = r % KH;  const int kd = r / KH;
    int id = od * SD - PD + kd;
    int ih = oh * SH - PH + kh;
    int iw = ow * SW - PW + kw;
    bool inb = posOK && id >= 0 && id < DIN && ih >= 0 && ih < HIN && iw >= 0 && iw < WIN;
    int ipos = inb ? (((ob * DIN + id) * HIN + ih) * WIN + iw) : 0;
    const bf16* abase = in + (size_t)ipos * CSIN;
    if (inb) __builtin_prefetch(abase + CSIN, 0, 0);  // next tap is +1 voxel in W

#pragma unroll
    for (int ks = 0; ks < KSL; ++ks) {
      // A 16x32 bf16 layout: lanes 0-15 K={0..7,16..23}; lanes 16-31 K={8..15,24..31}
      const int kc0 = ks * 32 + half * 8;
      const int kc1 = ks * 32 + 16 + half * 8;
      ABfrag a;
      a.u[0] = (inb && (kc0 + 8) <= CSIN) ? *(const u32x4*)(abase + kc0) : z4;
      a.u[1] = (inb && (kc1 + 8) <= CSIN) ? *(const u32x4*)(abase + kc1) : z4;
#pragma unroll
      for (int nt = 0; nt < NT; ++nt) {
        // B 32x16: lanes 0-15 col N=lane K={0..7,16..23}; lanes 16-31 col N=lane-16 K={8..15,24..31}
        ABfrag b;
        const u32x4* bp =
            (const u32x4*)(sB + (((nt * TAPS + tp) * KSL + ks) * 16 + m) * 32 + half * 8);
        b.u[0] = bp[0];
        b.u[1] = bp[2];
        acc[nt] = __builtin_amdgcn_wmma_f32_16x16x32_bf16(false, a.v, false, b.v,
                                                          (short)0, acc[nt], false, false);
      }
    }
  }

  // Mask is per-position (N-independent): load once
  float mk[8];
#pragma unroll
  for (int r8 = 0; r8 < 8; ++r8) {
    int pos = mBase + half * 8 + r8;
    mk[r8] = (pos < NUMM) ? maskOut[pos] : 0.f;
  }

  // Fused BN + ReLU + mask.  D layout: VGPR r -> M = r + half*8, N = lane&15
#pragma unroll
  for (int nt = 0; nt < NT; ++nt) {
    const int n = (ng * NT + nt) * 16 + m;
    const float sc = bng[n] * rsqrtf(bnv[n] + 1e-3f);
    const float mu = bnm[n];
    const float bi = bnb[n];
#pragma unroll
    for (int r8 = 0; r8 < 8; ++r8) {
      int pos = mBase + half * 8 + r8;
      if (pos >= NUMM) continue;
      float y = (acc[nt][r8] - mu) * sc + bi;
      y = fmaxf(y, 0.f) * mk[r8];
      if (FINAL) {
        int w2 = pos % WOUT; int t2 = pos / WOUT;
        int h2 = t2 % HOUT; t2 /= HOUT;
        int d2 = t2 % DOUT; int b2 = t2 / DOUT;
        fout[(((size_t)b2 * CSOUT + n) * DOUT + d2) * (HOUT * WOUT) + h2 * WOUT + w2] = y;
      } else {
        out[(size_t)pos * CSOUT + n] = (bf16)y;
      }
    }
  }
}

// ---------------------------------------------------------------------------
// Host helpers
// ---------------------------------------------------------------------------
template <int DIN, int HIN, int WIN, int DOUT, int HOUT, int WOUT,
          int CSIN, int CSOUT, int KSL, int NT,
          int KD, int KH, int KW, int SD, int SH, int SW,
          int PD, int PH, int PW, int FINAL>
static void run_conv(const bf16* in, bf16* out, float* fout, const bf16* wp,
                     const float* g, const float* b, const float* mn, const float* vr,
                     const float* mask, hipStream_t stream) {
  constexpr int NUMM = 2 * DOUT * HOUT * WOUT;
  dim3 grid((NUMM + 16 * WAVES - 1) / (16 * WAVES), CSOUT / (16 * NT));
  conv_wmma_kernel<DIN, HIN, WIN, DOUT, HOUT, WOUT, CSIN, CSOUT, KSL, NT,
                   KD, KH, KW, SD, SH, SW, PD, PH, PW, FINAL>
      <<<grid, TPB, 0, stream>>>(in, out, fout, wp, g, b, mn, vr, mask);
}

extern "C" void kernel_launch(void* const* d_in, const int* in_sizes, int n_in,
                              void* d_out, int out_size, void* d_ws, size_t ws_size,
                              hipStream_t stream) {
  const float* vf = (const float*)d_in[0];
  const int*   vc = (const int*)d_in[1];
  auto Wp = [&](int i) { return (const float*)d_in[2 + 5 * i]; };
  auto Bp = [&](int i, int j) { return (const float*)d_in[2 + 5 * i + 1 + j]; };

  char* ws = (char*)d_ws;
  size_t off = 0;
  auto alloc = [&](size_t bytes) -> char* {
    char* p = ws + off;
    off += (bytes + 255) & ~(size_t)255;
    return p;
  };

  const int vox0 = 2 * 33 * 160 * 160;
  const int vox1 = 2 * 17 * 80 * 80;
  const int vox2 = 2 * 9 * 40 * 40;
  const int vox3 = 2 * 4 * 20 * 20;
  const int vox4 = 2 * 1 * 20 * 20;

  float* m0 = (float*)alloc((size_t)vox0 * 4);
  float* m1 = (float*)alloc((size_t)vox1 * 4);
  float* m2 = (float*)alloc((size_t)vox2 * 4);
  float* m3 = (float*)alloc((size_t)vox3 * 4);
  float* m4 = (float*)alloc((size_t)vox4 * 4);

  bf16* x0  = (bf16*)alloc((size_t)vox0 * 16 * 2);
  bf16* xin = (bf16*)alloc((size_t)vox0 * 16 * 2);
  bf16* xc1 = (bf16*)alloc((size_t)vox0 * 16 * 2);
  bf16* a1  = (bf16*)alloc((size_t)vox1 * 32 * 2);
  bf16* b1  = (bf16*)alloc((size_t)vox1 * 32 * 2);
  bf16* a2  = (bf16*)alloc((size_t)vox2 * 64 * 2);
  bf16* b2  = (bf16*)alloc((size_t)vox2 * 64 * 2);
  bf16* a3  = (bf16*)alloc((size_t)vox3 * 64 * 2);
  bf16* b3  = (bf16*)alloc((size_t)vox3 * 64 * 2);

  auto wbytes = [](int nt, int taps, int ks) { return (size_t)nt * taps * ks * 512 * 2; };
  bf16* w_in  = (bf16*)alloc(wbytes(1, 27, 1));
  bf16* w_c1  = (bf16*)alloc(wbytes(1, 27, 1));
  bf16* w_2a  = (bf16*)alloc(wbytes(2, 27, 1));
  bf16* w_2b  = (bf16*)alloc(wbytes(2, 27, 1));
  bf16* w_2c  = (bf16*)alloc(wbytes(2, 27, 1));
  bf16* w_3a  = (bf16*)alloc(wbytes(4, 27, 1));
  bf16* w_3b  = (bf16*)alloc(wbytes(4, 27, 2));
  bf16* w_3c  = (bf16*)alloc(wbytes(4, 27, 2));
  bf16* w_4a  = (bf16*)alloc(wbytes(4, 27, 2));
  bf16* w_4b  = (bf16*)alloc(wbytes(4, 27, 2));
  bf16* w_4c  = (bf16*)alloc(wbytes(4, 27, 2));
  bf16* w_out = (bf16*)alloc(wbytes(8, 3, 2));

  // Zero-init only what is read-before-write
  hipMemsetAsync(x0, 0, (size_t)vox0 * 16 * 2, stream);
  hipMemsetAsync(m0, 0, (size_t)vox0 * 4, stream);

  // Scatter voxels
  const int N_ = in_sizes[0] / 4;
  scatter_kernel<<<(N_ + 255) / 256, 256, 0, stream>>>(vf, vc, x0, m0, N_);

  // Mask pyramid
  mask_down_kernel<33, 160, 160, 17, 80, 80, 3, 3, 3, 2, 2, 2, 1, 1, 1>
      <<<(2 * 17 * 80 * 80 + 255) / 256, 256, 0, stream>>>(m0, m1);
  mask_down_kernel<17, 80, 80, 9, 40, 40, 3, 3, 3, 2, 2, 2, 1, 1, 1>
      <<<(2 * 9 * 40 * 40 + 255) / 256, 256, 0, stream>>>(m1, m2);
  mask_down_kernel<9, 40, 40, 4, 20, 20, 3, 3, 3, 2, 2, 2, 0, 1, 1>
      <<<(2 * 4 * 20 * 20 + 255) / 256, 256, 0, stream>>>(m2, m3);
  mask_down_kernel<4, 20, 20, 1, 20, 20, 3, 1, 1, 2, 1, 1, 0, 0, 0>
      <<<(2 * 1 * 20 * 20 + 255) / 256, 256, 0, stream>>>(m3, m4);

  // Weight prep (only live layers; rest of the reference graph is dead)
  auto blocks = [](int total) { return (total + 255) / 256; };
  prep_w_kernel<16, 4, 3, 3, 3, 1><<<blocks(1 * 27 * 1 * 512), 256, 0, stream>>>(Wp(0), w_in);
  prep_w_kernel<16, 16, 3, 3, 3, 1><<<blocks(1 * 27 * 1 * 512), 256, 0, stream>>>(Wp(1), w_c1);
  prep_w_kernel<32, 16, 3, 3, 3, 1><<<blocks(2 * 27 * 1 * 512), 256, 0, stream>>>(Wp(9), w_2a);
  prep_w_kernel<32, 32, 3, 3, 3, 1><<<blocks(2 * 27 * 1 * 512), 256, 0, stream>>>(Wp(10), w_2b);
  prep_w_kernel<32, 32, 3, 3, 3, 1><<<blocks(2 * 27 * 1 * 512), 256, 0, stream>>>(Wp(11), w_2c);
  prep_w_kernel<64, 32, 3, 3, 3, 1><<<blocks(4 * 27 * 1 * 512), 256, 0, stream>>>(Wp(12), w_3a);
  prep_w_kernel<64, 64, 3, 3, 3, 2><<<blocks(4 * 27 * 2 * 512), 256, 0, stream>>>(Wp(13), w_3b);
  prep_w_kernel<64, 64, 3, 3, 3, 2><<<blocks(4 * 27 * 2 * 512), 256, 0, stream>>>(Wp(14), w_3c);
  prep_w_kernel<64, 64, 3, 3, 3, 2><<<blocks(4 * 27 * 2 * 512), 256, 0, stream>>>(Wp(15), w_4a);
  prep_w_kernel<64, 64, 3, 3, 3, 2><<<blocks(4 * 27 * 2 * 512), 256, 0, stream>>>(Wp(16), w_4b);
  prep_w_kernel<64, 64, 3, 3, 3, 2><<<blocks(4 * 27 * 2 * 512), 256, 0, stream>>>(Wp(17), w_4c);
  prep_w_kernel<128, 64, 3, 1, 1, 2><<<blocks(8 * 3 * 2 * 512), 256, 0, stream>>>(Wp(18), w_out);

  // Live network path.  <Din,Hin,Win, Dout,Hout,Wout, CsIn,CsOut, KSL,NT, K..., S..., P..., FINAL>
  run_conv<33,160,160, 33,160,160, 16, 16, 1,1, 3,3,3, 1,1,1, 1,1,1, 0>(
      x0, xin, nullptr, w_in, Bp(0,0), Bp(0,1), Bp(0,2), Bp(0,3), m0, stream);
  run_conv<33,160,160, 33,160,160, 16, 16, 1,1, 3,3,3, 1,1,1, 1,1,1, 0>(
      xin, xc1, nullptr, w_c1, Bp(1,0), Bp(1,1), Bp(1,2), Bp(1,3), m0, stream);
  run_conv<33,160,160, 17,80,80, 16, 32, 1,2, 3,3,3, 2,2,2, 1,1,1, 0>(
      xc1, a1, nullptr, w_2a, Bp(9,0), Bp(9,1), Bp(9,2), Bp(9,3), m1, stream);
  run_conv<17,80,80, 17,80,80, 32, 32, 1,2, 3,3,3, 1,1,1, 1,1,1, 0>(
      a1, b1, nullptr, w_2b, Bp(10,0), Bp(10,1), Bp(10,2), Bp(10,3), m1, stream);
  run_conv<17,80,80, 17,80,80, 32, 32, 1,2, 3,3,3, 1,1,1, 1,1,1, 0>(
      b1, a1, nullptr, w_2c, Bp(11,0), Bp(11,1), Bp(11,2), Bp(11,3), m1, stream);
  run_conv<17,80,80, 9,40,40, 32, 64, 1,4, 3,3,3, 2,2,2, 1,1,1, 0>(
      a1, a2, nullptr, w_3a, Bp(12,0), Bp(12,1), Bp(12,2), Bp(12,3), m2, stream);
  run_conv<9,40,40, 9,40,40, 64, 64, 2,2, 3,3,3, 1,1,1, 1,1,1, 0>(
      a2, b2, nullptr, w_3b, Bp(13,0), Bp(13,1), Bp(13,2), Bp(13,3), m2, stream);
  run_conv<9,40,40, 9,40,40, 64, 64, 2,2, 3,3,3, 1,1,1, 1,1,1, 0>(
      b2, a2, nullptr, w_3c, Bp(14,0), Bp(14,1), Bp(14,2), Bp(14,3), m2, stream);
  run_conv<9,40,40, 4,20,20, 64, 64, 2,2, 3,3,3, 2,2,2, 0,1,1, 0>(
      a2, a3, nullptr, w_4a, Bp(15,0), Bp(15,1), Bp(15,2), Bp(15,3), m3, stream);
  run_conv<4,20,20, 4,20,20, 64, 64, 2,2, 3,3,3, 1,1,1, 1,1,1, 0>(
      a3, b3, nullptr, w_4b, Bp(16,0), Bp(16,1), Bp(16,2), Bp(16,3), m3, stream);
  run_conv<4,20,20, 4,20,20, 64, 64, 2,2, 3,3,3, 1,1,1, 1,1,1, 0>(
      b3, a3, nullptr, w_4c, Bp(17,0), Bp(17,1), Bp(17,2), Bp(17,3), m3, stream);
  run_conv<4,20,20, 1,20,20, 64, 128, 2,8, 3,1,1, 2,1,1, 0,0,0, 1>(
      a3, nullptr, (float*)d_out, w_out, Bp(18,0), Bp(18,1), Bp(18,2), Bp(18,3), m4, stream);

  (void)n_in; (void)out_size; (void)ws_size;
}
